// WeaklySpatialEmbLoss_73074573574542
// MI455X (gfx1250) — matrix operand order repaired and checked by reading, not compiled, not run on hardware
//
#include <hip/hip_runtime.h>

typedef __attribute__((ext_vector_type(16))) __bf16 v16bf;
typedef __attribute__((ext_vector_type(8)))  float  v8f;
typedef __attribute__((ext_vector_type(16))) unsigned short v16u;

union V16 { v16u u; v16bf b; };

#define Bn   2
#define Nn   8
#define Hh   512
#define Ww   1024
#define HWn  (Hh * Ww)        // 524288
#define SEGS 16
#define TILESZ 4096
#define TILES  128            // HWn / TILESZ
#define NBINS  256            // 8-bit radix digits

// float-index offsets inside accumulator region (2KB, zeroed per launch)
#define ACC_CNT    0
#define ACC_SUMX   16
#define ACC_SUMY   32
#define ACC_SUMS0  48
#define ACC_SUMS1  64
#define ACC_CX     80
#define ACC_CY     96
#define ACC_SM0    112
#define ACC_SM1    128
#define ACC_SE0    144
#define ACC_SE1    160
#define ACC_VALID  176
#define ACC_VAR    192
#define ACC_SF     208
#define ACC_LOV    224
#define ACC_SEEDBG 240

__device__ __forceinline__ unsigned short f2bfu(float f) {
    unsigned x = __float_as_uint(f);
    return (unsigned short)((x + 0x7FFFu + ((x >> 16) & 1u)) >> 16);
}

__device__ __forceinline__ float sigmoidf_(float x) {
    return 1.0f / (1.0f + expf(-x));
}

__device__ __forceinline__ unsigned rdig8(unsigned key, int shift) {
    return ((~key) >> shift) & 255u;   // complemented digit -> descending sort
}

// ---------------------------------------------------------------------------
// k0: precompute spatial embedding planes (tanh + coord grid), f32
//     emb[(b*2+0)*HW + pix] = tanh(pred[b,0]) + x ; [..+1..] = tanh(pred[b,1]) + y
// ---------------------------------------------------------------------------
__global__ __launch_bounds__(256)
void k0_emb(const float* __restrict__ pred, float* __restrict__ emb) {
    const int total = Bn * HWn;
    for (int i = blockIdx.x * 256 + threadIdx.x; i < total; i += gridDim.x * 256) {
        const int b = i / HWn, pix = i - b * HWn;
        const float* predb = pred + (size_t)b * 12 * HWn;
        float ex = tanhf(predb[pix])       + (2.0f / 2047.0f) * (float)(pix & (Ww - 1));
        float ey = tanhf(predb[HWn + pix]) + (1.0f / 1023.0f) * (float)(pix >> 10);
        emb[(size_t)(b * 2 + 0) * HWn + pix] = ex;
        emb[(size_t)(b * 2 + 1) * HWn + pix] = ey;
    }
}

// ---------------------------------------------------------------------------
// k1: WMMA reduction  C[channel][instance] += sum_pixels chan * mask
//     rows: 0=emb_x 1=emb_y 2=sigma0 3=sigma1 4=ones ; cols: instances 0..7
//     plus seed_bg reduction (phase 2).
// ---------------------------------------------------------------------------
__global__ __launch_bounds__(256)
void k1_stats(const float* __restrict__ pred, const float* __restrict__ emb,
              const int* __restrict__ masks, const int* __restrict__ bbox,
              float* acc) {
    const int b    = blockIdx.x >> 6;     // 64 blocks per batch
    const int blk  = blockIdx.x & 63;
    const int lane = threadIdx.x & 31;
    const int wv   = threadIdx.x >> 5;
    const int gw   = blk * 8 + wv;        // 0..511 waves per batch
    const int ch   = lane & 15;           // A row / B col
    const int lo   = (lane < 16) ? 0 : 8; // A K-offset within chunk
    const int koff = (lane < 16) ? 0 : 16;// unused: B K-offset is lo, as for A

    const float* predb = pred + (size_t)b * 12 * HWn;
    const int*   maskn = masks + ((size_t)(b * 8 + (ch < 8 ? ch : 0))) * HWn;
    // channel source plane: emb planes for rows 0/1, raw sigma planes for 2/3
    const float* cp = (ch < 2) ? (emb + (size_t)(b * 2 + ch) * HWn)
                               : (predb + (size_t)(ch < 4 ? ch : 2) * HWn);

    v8f c;
    #pragma unroll
    for (int e = 0; e < 8; ++e) c[e] = 0.0f;

    for (int chunk = gw; chunk < HWn / 32; chunk += 512) {
        const int p0 = chunk * 32;
        V16 A, Bm;
        // ---- A operand: lane = channel row, identity K->pixel map ----
        if (ch < 4) {
            float f[16];
            #pragma unroll
            for (int e = 0; e < 8; ++e) f[e]     = cp[p0 + lo + e];
            #pragma unroll
            for (int e = 0; e < 8; ++e) f[8 + e] = cp[p0 + 16 + lo + e];
            #pragma unroll
            for (int e = 0; e < 16; ++e) A.u[e] = f2bfu(f[e]);
        } else {
            const unsigned short fv = (ch == 4) ? 0x3F80 : 0;  // ones row / zero rows
            #pragma unroll
            for (int e = 0; e < 16; ++e) A.u[e] = fv;
        }
        // ---- B operand: lane = instance column ----
        if (ch < 8) {
            #pragma unroll
            for (int e = 0; e < 16; ++e)
                Bm.u[e] = (maskn[p0 + (e < 8 ? lo + e : 16 + lo + (e - 8))] > 0) ? 0x3F80 : 0;
        } else {
            #pragma unroll
            for (int e = 0; e < 16; ++e) Bm.u[e] = 0;
        }
        c = __builtin_amdgcn_wmma_f32_16x16x32_bf16(false, A.b, false, Bm.b,
                                                    (short)0, c, false, false);
    }
    if (lane < 8) {
        const int idx = b * 8 + lane;
        atomicAdd(&acc[ACC_SUMX  + idx], c[0]);
        atomicAdd(&acc[ACC_SUMY  + idx], c[1]);
        atomicAdd(&acc[ACC_SUMS0 + idx], c[2]);
        atomicAdd(&acc[ACC_SUMS1 + idx], c[3]);
        atomicAdd(&acc[ACC_CNT   + idx], c[4]);
    }

    // ---- phase 2: seed_bg[b] = sum over bg pixels of sigmoid(seed)^2 ----
    const int* bb   = bbox + (size_t)b * 9 * HWn;
    const int  ltid = blk * 256 + threadIdx.x;  // 0..16383
    float sbg = 0.0f;
    for (int pix = ltid; pix < HWn; pix += 64 * 256) {
        #pragma unroll
        for (int k = 0; k < 8; ++k) {
            if (bb[(size_t)(1 + k) * HWn + pix] == 0) {
                float s = sigmoidf_(predb[(size_t)(4 + k) * HWn + pix]);
                sbg += s * s;
            }
        }
    }
    for (int off = 16; off; off >>= 1) sbg += __shfl_down(sbg, off, 32);
    if (lane == 0) atomicAdd(&acc[ACC_SEEDBG + b], sbg);
}

// ---------------------------------------------------------------------------
// k2: tiny per-instance finalize: center, s_mean, s_exp, valid
// ---------------------------------------------------------------------------
__global__ void k2_finalize(float* acc) {
    int t = threadIdx.x;
    if (t < 16) {
        float c  = acc[ACC_CNT + t];
        float cs = fmaxf(c, 1.0f);
        acc[ACC_CX + t] = acc[ACC_SUMX + t] / cs;
        acc[ACC_CY + t] = acc[ACC_SUMY + t] / cs;
        float sm0 = acc[ACC_SUMS0 + t] / cs;
        float sm1 = acc[ACC_SUMS1 + t] / cs;
        acc[ACC_SM0 + t] = sm0;
        acc[ACC_SM1 + t] = sm1;
        acc[ACC_SE0 + t] = expf(sm0 * 10.0f);
        acc[ACC_SE1 + t] = expf(sm1 * 10.0f);
        acc[ACC_VALID + t] = (c >= 128.0f) ? 1.0f : 0.0f;
    }
}

// ---------------------------------------------------------------------------
// k3: per-pixel pass: dist -> sort keys, var_inst & seed_fg reductions
//     key = (bits(error) << 1) | label   (error in [0,2], sign bit 0)
// ---------------------------------------------------------------------------
__global__ __launch_bounds__(256)
void k3_pixel(const float* __restrict__ pred, const float* __restrict__ emb,
              const int* __restrict__ masks, const int* __restrict__ cls_ids,
              float* acc, unsigned* __restrict__ keyA) {
    const int b   = blockIdx.x >> 9;   // 512 blocks per batch
    const int blk = blockIdx.x & 511;
    __shared__ float s_cx[8], s_cy[8], s_se0[8], s_se1[8], s_sm0[8], s_sm1[8];
    __shared__ int   s_cls[8];
    if (threadIdx.x < 8) {
        int idx = b * 8 + threadIdx.x;
        s_cx[threadIdx.x]  = acc[ACC_CX  + idx];
        s_cy[threadIdx.x]  = acc[ACC_CY  + idx];
        s_se0[threadIdx.x] = acc[ACC_SE0 + idx];
        s_se1[threadIdx.x] = acc[ACC_SE1 + idx];
        s_sm0[threadIdx.x] = acc[ACC_SM0 + idx];
        s_sm1[threadIdx.x] = acc[ACC_SM1 + idx];
        s_cls[threadIdx.x] = cls_ids[idx] & 7;
    }
    __syncthreads();

    const float* predb = pred  + (size_t)b * 12 * HWn;
    const float* embb  = emb   + (size_t)b * 2 * HWn;
    const int*   maskb = masks + (size_t)b * 8 * HWn;
    float vn[8], sf[8];
    #pragma unroll
    for (int n2 = 0; n2 < 8; ++n2) { vn[n2] = 0.0f; sf[n2] = 0.0f; }

    for (int pix = blk * 256 + threadIdx.x; pix < HWn; pix += 512 * 256) {
        float ex = embb[pix];
        float ey = embb[HWn + pix];
        float s0 = predb[2 * (size_t)HWn + pix];
        float s1 = predb[3 * (size_t)HWn + pix];
        #pragma unroll
        for (int n2 = 0; n2 < 8; ++n2) {
            int   mlab = (maskb[(size_t)n2 * HWn + pix] > 0) ? 1 : 0;
            float dx = ex - s_cx[n2], dy = ey - s_cy[n2];
            float dist = expf(-(dx * dx * s_se0[n2] + dy * dy * s_se1[n2]));
            float err  = mlab ? (2.0f - 2.0f * dist) : (2.0f * dist);
            err = fmaxf(err, 0.0f);
            keyA[((size_t)(b * 8 + n2)) * HWn + pix] =
                (__float_as_uint(err) << 1) | (unsigned)mlab;
            if (mlab) {
                float d0 = s0 - s_sm0[n2], d1 = s1 - s_sm1[n2];
                vn[n2] += d0 * d0 + d1 * d1;
                float sc = sigmoidf_(predb[(size_t)(4 + s_cls[n2]) * HWn + pix]);
                float dd = sc - dist;
                sf[n2] += dd * dd;
            }
        }
    }
    const int lane = threadIdx.x & 31, wv = threadIdx.x >> 5;
    #pragma unroll
    for (int n2 = 0; n2 < 8; ++n2) {
        for (int off = 16; off; off >>= 1) {
            vn[n2] += __shfl_down(vn[n2], off, 32);
            sf[n2] += __shfl_down(sf[n2], off, 32);
        }
    }
    __shared__ float red[8][16];
    if (lane == 0) {
        #pragma unroll
        for (int n2 = 0; n2 < 8; ++n2) { red[wv][n2] = vn[n2]; red[wv][8 + n2] = sf[n2]; }
    }
    __syncthreads();
    if (threadIdx.x < 16) {
        float s = 0.0f;
        #pragma unroll
        for (int w = 0; w < 8; ++w) s += red[w][threadIdx.x];
        if (threadIdx.x < 8) atomicAdd(&acc[ACC_VAR + b * 8 + threadIdx.x], s);
        else                 atomicAdd(&acc[ACC_SF  + b * 8 + (threadIdx.x - 8)], s);
    }
}

// ---------------------------------------------------------------------------
// Segmented 8-bit LSD radix sort, 4 passes, descending via complemented digit.
// Tile = 4096 keys, 256 threads (8 waves); each wave owns a CONTIGUOUS 512-key
// sub-block so (wave, iter, lane) order == memory order -> stable ranking.
// ---------------------------------------------------------------------------
__global__ __launch_bounds__(256)
void k4_hist(const unsigned* __restrict__ in, unsigned* __restrict__ hist, int shift) {
    const int seg = blockIdx.x >> 7, tile = blockIdx.x & 127;
    const int t = threadIdx.x;
    __shared__ unsigned h[NBINS];
    h[t] = 0;
    __syncthreads();
    const size_t base = (size_t)seg * HWn + (size_t)tile * TILESZ + (size_t)t * 16;
    #pragma unroll
    for (int j = 0; j < 16; ++j)
        atomicAdd(&h[rdig8(in[base + j], shift)], 1u);
    __syncthreads();
    hist[(size_t)blockIdx.x * NBINS + t] = h[t];
}

__global__ __launch_bounds__(256)
void k5_scan(const unsigned* __restrict__ hist, unsigned* __restrict__ basebuf) {
    const int seg = blockIdx.x;
    const int d = threadIdx.x;            // digit 0..255
    __shared__ unsigned scn[NBINS];
    unsigned run = 0;
    for (int tile = 0; tile < TILES; ++tile) {
        size_t idx = ((size_t)(seg * TILES + tile)) * NBINS + d;
        unsigned v = hist[idx];
        basebuf[idx] = run;               // per-digit tile prefix
        run += v;
    }
    const unsigned tot = run;
    scn[d] = tot;
    __syncthreads();
    for (int off = 1; off < NBINS; off <<= 1) {   // inclusive scan over digits
        unsigned v = (d >= off) ? scn[d - off] : 0;
        __syncthreads();
        scn[d] += v;
        __syncthreads();
    }
    const unsigned dbase = scn[d] - tot + (unsigned)(seg * HWn);  // exclusive + seg base
    for (int tile = 0; tile < TILES; ++tile)
        basebuf[((size_t)(seg * TILES + tile)) * NBINS + d] += dbase;
}

__global__ __launch_bounds__(256)
void k6_scatter(const unsigned* __restrict__ in, unsigned* __restrict__ out,
                const unsigned* __restrict__ basebuf, int shift) {
    const int seg = blockIdx.x >> 7, tile = blockIdx.x & 127;
    const int t = threadIdx.x;
    const int lane = t & 31, wv = t >> 5;
    __shared__ unsigned wh[8 * NBINS];    // per-wave digit counts -> exclusive
    __shared__ unsigned sbase[NBINS];
    #pragma unroll
    for (int k = 0; k < 8; ++k) wh[k * NBINS + t] = 0;
    sbase[t] = basebuf[(size_t)blockIdx.x * NBINS + t];
    __syncthreads();

    const unsigned ltmask = (1u << lane) - 1u;
    const size_t base = (size_t)seg * HWn + (size_t)tile * TILESZ + (size_t)wv * 512;
    unsigned keys[16], dr[16];
    #pragma unroll
    for (int j = 0; j < 16; ++j) {
        const unsigned key = in[base + j * 32 + lane];
        keys[j] = key;
        const unsigned d = rdig8(key, shift);
        // wave32 multi-split: lanes sharing this digit
        unsigned mask = 0xFFFFFFFFu;
        #pragma unroll
        for (int k2 = 0; k2 < 8; ++k2) {
            unsigned bal = (unsigned)__ballot((int)((d >> k2) & 1u));
            mask &= ((d >> k2) & 1u) ? bal : ~bal;
        }
        const unsigned before = wh[wv * NBINS + d];        // all lanes read first
        const unsigned rank = before + __popc(mask & ltmask);
        if ((mask & ltmask) == 0)                          // lowest lane of group
            wh[wv * NBINS + d] = before + __popc(mask);
        dr[j] = (d << 16) | rank;                          // rank < 4096
    }
    __syncthreads();
    // cross-wave exclusive scan per digit (thread t owns digit t)
    unsigned run = 0;
    #pragma unroll
    for (int w = 0; w < 8; ++w) {
        unsigned v = wh[w * NBINS + t];
        wh[w * NBINS + t] = run;
        run += v;
    }
    __syncthreads();
    #pragma unroll
    for (int j = 0; j < 16; ++j) {
        const unsigned d = dr[j] >> 16;
        const unsigned pos = sbase[d] + wh[wv * NBINS + d] + (dr[j] & 0xFFFFu);
        out[pos] = keys[j];
    }
}

// ---------------------------------------------------------------------------
// k7: Lovasz hinge over sorted keys (block-parallel cumsum per segment)
// ---------------------------------------------------------------------------
__global__ __launch_bounds__(256)
void k7_lovasz(const unsigned* __restrict__ keys, float* acc) {
    const int seg = blockIdx.x;
    const int t = threadIdx.x;
    const float gts = acc[ACC_CNT + seg];
    const unsigned* kp = keys + (size_t)seg * HWn;
    __shared__ int scn[256];
    __shared__ float red[8];
    int carry = 0;
    float loss = 0.0f;
    for (int tile = 0; tile < TILES; ++tile) {
        const int i0 = tile * TILESZ + t * 16;
        if (tile + 1 < TILES)
            __builtin_prefetch(kp + i0 + TILESZ, 0, 1);   // global_prefetch_b8
        unsigned kk[16];
        int ls = 0;
        #pragma unroll
        for (int j = 0; j < 16; ++j) { kk[j] = kp[i0 + j]; ls += (int)(kk[j] & 1u); }
        scn[t] = ls;
        __syncthreads();
        for (int off = 1; off < 256; off <<= 1) {
            int v = (t >= off) ? scn[t - off] : 0;
            __syncthreads();
            scn[t] += v;
            __syncthreads();
        }
        const int incl = scn[t];
        const int total = scn[255];
        int cs = carry + incl - ls;         // cumsum before this thread's items
        float prevjac;
        if (i0 == 0) prevjac = 0.0f;
        else {
            float csf = (float)cs, fi0 = (float)i0;
            prevjac = 1.0f - (gts - csf) / (gts + fi0 - csf);
        }
        #pragma unroll
        for (int j = 0; j < 16; ++j) {
            const unsigned key = kk[j];
            const int lab = (int)(key & 1u);
            const float err = __uint_as_float(key >> 1);
            cs += lab;
            float csf = (float)cs;
            float fi1 = (float)(i0 + j + 1);
            float jac = 1.0f - (gts - csf) / (gts + fi1 - csf);
            loss += err * (jac - prevjac);
            prevjac = jac;
        }
        carry += total;
        __syncthreads();                    // protect scn[] reuse next tile
    }
    const int lane = t & 31, wv = t >> 5;
    for (int off = 16; off; off >>= 1) loss += __shfl_down(loss, off, 32);
    if (lane == 0) red[wv] = loss;
    __syncthreads();
    if (t == 0) {
        float s = 0.0f;
        #pragma unroll
        for (int w = 0; w < 8; ++w) s += red[w];
        acc[ACC_LOV + seg] = s;
    }
}

// ---------------------------------------------------------------------------
// k8: final scalar combine
// ---------------------------------------------------------------------------
__global__ void k8_final(const float* __restrict__ acc, const int* __restrict__ cls_ids,
                         float* __restrict__ out) {
    if (threadIdx.x != 0 || blockIdx.x != 0) return;
    const float FW[8] = {10.0f, 10.0f, 10.0f, 40.0f, 80.0f, 100.0f, 60.0f, 20.0f};
    float total = 0.0f;
    for (int b = 0; b < Bn; ++b) {
        float obj = 0.0f, instl = 0.0f, varl = 0.0f, sfsum = 0.0f;
        for (int n2 = 0; n2 < 8; ++n2) {
            int idx = b * 8 + n2;
            float valid = acc[ACC_VALID + idx];
            float cs = fmaxf(acc[ACC_CNT + idx], 1.0f);
            obj   += valid;
            instl += acc[ACC_LOV + idx] * valid;
            varl  += (acc[ACC_VAR + idx] / (2.0f * cs)) * valid;
            sfsum += FW[cls_ids[idx] & 7] * acc[ACC_SF + idx] * valid;
        }
        float objs   = fmaxf(obj, 1.0f);
        float seed_l = (acc[ACC_SEEDBG + b] + sfsum) / (float)HWn;
        total += (instl / objs) + 10.0f * (varl / objs) + seed_l;
    }
    out[0] = total * 0.5f;  // mean over B=2
}

// ---------------------------------------------------------------------------
extern "C" void kernel_launch(void* const* d_in, const int* in_sizes, int n_in,
                              void* d_out, int out_size, void* d_ws, size_t ws_size,
                              hipStream_t stream) {
    (void)in_sizes; (void)n_in; (void)out_size; (void)ws_size;
    const float* pred  = (const float*)d_in[0];
    const int*   bbox  = (const int*)d_in[1];
    const int*   masks = (const int*)d_in[2];
    const int*   cls   = (const int*)d_in[3];

    char* ws = (char*)d_ws;
    const size_t HIST_SZ = (size_t)SEGS * TILES * NBINS * 4;        // 2 MB
    const size_t KEY_SZ  = (size_t)SEGS * HWn * 4;                  // 32 MB
    float*    acc     = (float*)ws;                                 // 2 KB
    unsigned* hist    = (unsigned*)(ws + 2048);
    unsigned* basebuf = (unsigned*)(ws + 2048 + HIST_SZ);
    unsigned* keyA    = (unsigned*)(ws + 2048 + 2 * HIST_SZ);
    unsigned* keyB    = (unsigned*)(ws + 2048 + 2 * HIST_SZ + KEY_SZ);
    float*    emb     = (float*)   (ws + 2048 + 2 * HIST_SZ + 2 * KEY_SZ);  // 8 MB

    hipMemsetAsync(acc, 0, 2048, stream);
    k0_emb<<<2048, 256, 0, stream>>>(pred, emb);
    k1_stats<<<Bn * 64, 256, 0, stream>>>(pred, emb, masks, bbox, acc);
    k2_finalize<<<1, 32, 0, stream>>>(acc);
    k3_pixel<<<Bn * 512, 256, 0, stream>>>(pred, emb, masks, cls, acc, keyA);
    for (int p = 0; p < 4; ++p) {                 // 4 passes of 8 bits
        unsigned* src = (p & 1) ? keyB : keyA;
        unsigned* dst = (p & 1) ? keyA : keyB;
        const int shift = p * 8;
        k4_hist   <<<SEGS * TILES, 256, 0, stream>>>(src, hist, shift);
        k5_scan   <<<SEGS, 256, 0, stream>>>(hist, basebuf);
        k6_scatter<<<SEGS * TILES, 256, 0, stream>>>(src, dst, basebuf, shift);
    }
    k7_lovasz<<<SEGS, 256, 0, stream>>>(keyA, acc);   // 4 passes end in keyA
    k8_final<<<1, 32, 0, stream>>>(acc, cls, (float*)d_out);
}
